// SAFTEncoder_51273319579763
// MI455X (gfx1250) — compile-verified, matrix operand order checked by reading, not run on hardware
//
#include <hip/hip_runtime.h>

// ---------------------------------------------------------------------------
// Types (plain ext_vectors so they can live in unions)
// ---------------------------------------------------------------------------
typedef __attribute__((ext_vector_type(16))) __bf16        v16bf;
typedef __attribute__((ext_vector_type(8)))  float         v8f;
typedef __attribute__((ext_vector_type(4)))  unsigned int  v4u;
typedef __attribute__((ext_vector_type(2)))  unsigned int  v2u;
typedef __attribute__((ext_vector_type(4)))  float         v4f;

union FragU { v4u u[2]; v16bf v; };

__device__ __forceinline__ unsigned short f2bf(float f) {
    unsigned int u = __float_as_uint(f);
    u += 0x7FFFu + ((u >> 16) & 1u);           // round-to-nearest-even
    return (unsigned short)(u >> 16);
}
__device__ __forceinline__ unsigned int pack2(float a, float b) {
    return (unsigned int)f2bf(a) | ((unsigned int)f2bf(b) << 16);
}

// ---------------------------------------------------------------------------
// Generic batched GEMM: C = act(A @ B + bias + resScale*Res)
//   A: [M,K] row-major (batch stride sA), B: [K,N] row-major (stride sB)
//   C/Res: [M,N] (stride sC).  BM=128, BN=64, BK=32; 8 waves (4x2), each wave
//   owns 2x2 WMMA 16x16 tiles.  fp32 in memory, bf16 WMMA with f32 accum.
// ---------------------------------------------------------------------------
#define LDST 40   // padded LDS row stride in halfwords (32 + 8)

template<int ACT>  // 0=none, 1=gelu(exact)
__global__ __launch_bounds__(256) void gemm_bf16_kernel(
    const float* __restrict__ A, const float* __restrict__ B,
    const float* __restrict__ bias, const float* __restrict__ Res, float resScale,
    float* __restrict__ C, int M, int N, int K,
    long long sA, long long sB, long long sC)
{
    __shared__ unsigned short As[128 * LDST];
    __shared__ unsigned short Bs[64 * LDST];

    const int t  = threadIdx.x;
    const int m0 = blockIdx.x * 128;
    const int n0 = blockIdx.y * 64;
    const long long zb = blockIdx.z;
    const float* Ab = A + zb * sA;
    const float* Bb = B + zb * sB;
    float*       Cb = C + zb * sC;
    const float* Rb = Res ? (Res + zb * sC) : nullptr;

    const int wave = t >> 5, lane = t & 31;
    const int wm = wave >> 1, wn = wave & 1;        // 4x2 wave grid
    const int lr = lane & 15, lh = lane >> 4;       // lane-row, lane-half
    const int klo = lh * 8;                         // K base for this half

    v8f acc[2][2];
    #pragma unroll
    for (int mi = 0; mi < 2; ++mi)
        #pragma unroll
        for (int ni = 0; ni < 2; ++ni)
            #pragma unroll
            for (int j = 0; j < 8; ++j) acc[mi][ni][j] = 0.f;

    for (int k0 = 0; k0 < K; k0 += 32) {
        // ---- stage A tile (128x32 fp32 -> bf16) ----
        #pragma unroll
        for (int i = 0; i < 4; ++i) {
            int lin = t + i * 256;                  // 1024 float4 slots
            int row = lin >> 3;
            int c4  = (lin & 7) << 2;
            v4f val = {0.f, 0.f, 0.f, 0.f};
            int gr = m0 + row;
            if (gr < M) val = *(const v4f*)&Ab[(long long)gr * K + k0 + c4];
            v2u pk = { pack2(val.x, val.y), pack2(val.z, val.w) };
            *(v2u*)&As[row * LDST + c4] = pk;
        }
        // ---- stage B tile transposed: Bs[n][k] (32x64 fp32 -> bf16) ----
        #pragma unroll
        for (int i = 0; i < 8; ++i) {
            int lin = t + i * 256;                  // 2048 scalars
            int kk = lin >> 6;
            int nn = lin & 63;
            Bs[nn * LDST + kk] = f2bf(Bb[(long long)(k0 + kk) * N + n0 + nn]);
        }
        __syncthreads();

        // ---- per-lane fragments (two ds_load_b128 each) ----
        FragU af[2], bfr[2];
        #pragma unroll
        for (int mi = 0; mi < 2; ++mi) {
            const unsigned short* pA = As + (wm * 32 + mi * 16 + lr) * LDST + klo;
            af[mi].u[0] = *(const v4u*)(pA);
            af[mi].u[1] = *(const v4u*)(pA + 16);
        }
        #pragma unroll
        for (int ni = 0; ni < 2; ++ni) {
            const unsigned short* pB = Bs + (wn * 32 + ni * 16 + lr) * LDST + klo;
            bfr[ni].u[0] = *(const v4u*)(pB);
            bfr[ni].u[1] = *(const v4u*)(pB + 16);
        }
        #pragma unroll
        for (int mi = 0; mi < 2; ++mi)
            #pragma unroll
            for (int ni = 0; ni < 2; ++ni)
                acc[mi][ni] = __builtin_amdgcn_wmma_f32_16x16x32_bf16(
                    false, af[mi].v, false, bfr[ni].v,
                    (short)0, acc[mi][ni], false, false);
        __syncthreads();
    }

    // ---- epilogue: bias + residual + activation ----
    #pragma unroll
    for (int mi = 0; mi < 2; ++mi) {
        #pragma unroll
        for (int ni = 0; ni < 2; ++ni) {
            int col = n0 + wn * 32 + ni * 16 + lr;
            float bv = bias ? bias[col] : 0.f;
            #pragma unroll
            for (int j = 0; j < 8; ++j) {
                int row = m0 + wm * 32 + mi * 16 + lh * 8 + j;
                if (row < M) {
                    float c = acc[mi][ni][j] + bv;
                    long long idx = (long long)row * N + col;
                    if (Rb) c += resScale * Rb[idx];
                    if (ACT == 1) c = 0.5f * c * (1.f + erff(c * 0.70710678118f));
                    Cb[idx] = c;
                }
            }
        }
    }
}

// ---------------------------------------------------------------------------
// Fused attention: one block per (sequence a, head).  S=32, dh=64, heads=4.
// ---------------------------------------------------------------------------
__global__ __launch_bounds__(256) void attn_kernel(
    const float* __restrict__ q, const float* __restrict__ k,
    const float* __restrict__ v, const float* __restrict__ mask,
    float* __restrict__ ctx, int layer0)
{
    __shared__ float qs[32 * 64], ks[32 * 64], vs[32 * 64];
    __shared__ float sc[32 * 33];
    const int blk = blockIdx.x;
    const int a = blk >> 2, head = blk & 3;
    const int t = threadIdx.x;
    const size_t base = (size_t)a * 32 * 256 + head * 64;

    #pragma unroll
    for (int i = 0; i < 2; ++i) {
        int lin = t + i * 256;                   // 512 float4 slots
        int row = lin >> 4;
        int c4  = (lin & 15) << 2;
        size_t g = base + (size_t)row * 256 + c4;
        *(v4f*)&qs[row * 64 + c4] = *(const v4f*)&q[g];
        *(v4f*)&ks[row * 64 + c4] = *(const v4f*)&k[g];
        *(v4f*)&vs[row * 64 + c4] = *(const v4f*)&v[g];
    }
    __syncthreads();

    #pragma unroll
    for (int i = 0; i < 4; ++i) {
        int lin = t + i * 256;                   // 1024 score entries
        int s = lin >> 5, tt = lin & 31;
        float d = 0.f;
        for (int e = 0; e < 64; ++e) d += qs[s * 64 + e] * ks[tt * 64 + e];
        float m = (layer0 && tt < 4) ? -10000.0f : mask[(size_t)a * 32 + tt];
        sc[s * 33 + tt] = d * 0.125f + m;
    }
    __syncthreads();

    if (t < 32) {   // softmax, one lane per row
        float mx = -3.4e38f;
        for (int j = 0; j < 32; ++j) mx = fmaxf(mx, sc[t * 33 + j]);
        float sum = 0.f;
        for (int j = 0; j < 32; ++j) {
            float e = __expf(sc[t * 33 + j] - mx);
            sc[t * 33 + j] = e; sum += e;
        }
        float inv = 1.f / sum;
        for (int j = 0; j < 32; ++j) sc[t * 33 + j] *= inv;
    }
    __syncthreads();

    #pragma unroll
    for (int i = 0; i < 8; ++i) {
        int lin = t + i * 256;                   // 2048 ctx entries
        int s = lin >> 6, d = lin & 63;
        float o = 0.f;
        for (int j = 0; j < 32; ++j) o += sc[s * 33 + j] * vs[j * 64 + d];
        ctx[base + (size_t)s * 256 + d] = o;
    }
}

// ---------------------------------------------------------------------------
// In-place LayerNorm over E=256; one wave per row (8 rows per block).
// ---------------------------------------------------------------------------
__global__ __launch_bounds__(256) void ln_kernel(
    float* __restrict__ X, const float* __restrict__ g,
    const float* __restrict__ b, int rows)
{
    int wave = threadIdx.x >> 5, lane = threadIdx.x & 31;
    int row = blockIdx.x * 8 + wave;
    if (row >= rows) return;
    size_t base = (size_t)row * 256;
    float vals[8], s = 0.f, s2 = 0.f;
    #pragma unroll
    for (int j = 0; j < 8; ++j) {
        float v = X[base + j * 32 + lane];
        vals[j] = v; s += v; s2 += v * v;
    }
    #pragma unroll
    for (int off = 16; off > 0; off >>= 1) {
        s  += __shfl_xor(s,  off);
        s2 += __shfl_xor(s2, off);
    }
    float mean = s * (1.f / 256.f);
    float var  = s2 * (1.f / 256.f) - mean * mean;
    float rs   = rsqrtf(var + 1e-12f);
    #pragma unroll
    for (int j = 0; j < 8; ++j) {
        int col = j * 32 + lane;
        X[base + col] = (vals[j] - mean) * rs * g[col] + b[col];
    }
}

// ---------------------------------------------------------------------------
// U[b,n,m] = (idx[b,n]==idx[b,m]) ? w[b,n]*w[b,m] : 0      (op: [B,N,2])
// ---------------------------------------------------------------------------
__global__ void sparse_prop_kernel(float* __restrict__ P, const float* __restrict__ op)
{
    int gidx = blockIdx.x * 256 + threadIdx.x;   // < 64*64*64
    int b = gidx >> 12, rem = gidx & 4095, n = rem >> 6, m = rem & 63;
    const float* on = op + ((size_t)b * 64 + n) * 2;
    const float* om = op + ((size_t)b * 64 + m) * 2;
    int in_ = (int)on[0], im = (int)om[0];
    P[gidx] = (in_ == im) ? on[1] * om[1] : 0.f;
}

// t0 = 0.5*emb (+ hsPrev[:, :, srow, :] if hsPrev)
__global__ void prop_init_kernel(float* __restrict__ t0, const float* __restrict__ emb,
                                 const float* __restrict__ hsPrev, int srow)
{
    int bn = blockIdx.x, e = threadIdx.x;
    float v = 0.5f * emb[(size_t)bn * 256 + e];
    if (hsPrev) v += hsPrev[((size_t)bn * 32 + srow) * 256 + e];
    t0[(size_t)bn * 256 + e] = v;
}

// hs[:, :, 0..3, :] = {qe, ke, de, ce}
__global__ void embed_set_kernel(float* __restrict__ hs, const float* __restrict__ qe,
                                 const float* __restrict__ ke, const float* __restrict__ de,
                                 const float* __restrict__ ce)
{
    int bn = blockIdx.x, e = threadIdx.x;
    size_t h = (size_t)bn * 32 * 256;
    size_t s = (size_t)bn * 256 + e;
    hs[h + 0 * 256 + e] = qe[s];
    hs[h + 1 * 256 + e] = ke[s];
    hs[h + 2 * 256 + e] = de[s];
    hs[h + 3 * 256 + e] = ce[s];
}

// agg: p = [l2norm(hs[:,0,4,:]), p1[:,0,:], p2[:,1,:]]; hs[:,0,4,:] =
//      l2norm(relu(p @ Wagg + bagg)).  One block per batch b, thread = out col.
__global__ __launch_bounds__(256) void agg_kernel(
    float* __restrict__ hs, const float* __restrict__ p1, const float* __restrict__ p2,
    const float* __restrict__ Wagg, const float* __restrict__ bagg)
{
    __shared__ float p[768];
    __shared__ float red[256];
    int b = blockIdx.x, e = threadIdx.x;
    size_t hIdx = (((size_t)b * 64 + 0) * 32 + 4) * 256 + e;
    float h4 = hs[hIdx];
    red[e] = h4 * h4; __syncthreads();
    for (int off = 128; off > 0; off >>= 1) {
        if (e < off) red[e] += red[e + off];
        __syncthreads();
    }
    float nrm = fmaxf(sqrtf(red[0]), 1e-12f);
    p[e]       = h4 / nrm;
    p[256 + e] = p1[((size_t)b * 64 + 0) * 256 + e];
    p[512 + e] = p2[((size_t)b * 64 + 1) * 256 + e];
    __syncthreads();
    float o = bagg[e];
    for (int j = 0; j < 768; ++j) o += p[j] * Wagg[j * 256 + e];
    o = fmaxf(o, 0.f);
    __syncthreads();
    red[e] = o * o; __syncthreads();
    for (int off = 128; off > 0; off >>= 1) {
        if (e < off) red[e] += red[e + off];
        __syncthreads();
    }
    float n2 = fmaxf(sqrtf(red[0]), 1e-12f);
    hs[hIdx] = o / n2;
}

// out[b,e] = 0.5*(mean_{t=4..31} hs1[b,0,t,e] + mean hs2[b,0,t,e])
__global__ void final_out_kernel(const float* __restrict__ hs1,
                                 const float* __restrict__ hs2, float* __restrict__ out)
{
    int b = blockIdx.x, e = threadIdx.x;
    size_t base = (size_t)b * 64 * 32 * 256;     // n = 0
    float s1 = 0.f, s2 = 0.f;
    for (int t = 4; t < 32; ++t) {
        s1 += hs1[base + t * 256 + e];
        s2 += hs2[base + t * 256 + e];
    }
    out[b * 256 + e] = 0.5f * (s1 + s2) * (1.f / 28.f);
}

// ---------------------------------------------------------------------------
// Orchestration
// ---------------------------------------------------------------------------
extern "C" void kernel_launch(void* const* d_in, const int* in_sizes, int n_in,
                              void* d_out, int out_size, void* d_ws, size_t ws_size,
                              hipStream_t stream)
{
    (void)in_sizes; (void)n_in; (void)out_size; (void)ws_size;

    const int B_ = 64, N_ = 64, S_ = 32, E_ = 256, FF_ = 1024, L_ = 2;
    const int AN = B_ * N_;                        // 4096
    const int MR = AN * S_;                        // 131072 rows

    const float* hidden = (const float*)d_in[1];
    const float* amask  = (const float*)d_in[2];
    const float* qe     = (const float*)d_in[3];
    const float* ke     = (const float*)d_in[4];
    const float* de     = (const float*)d_in[5];
    const float* ce     = (const float*)d_in[6];
    const float* uop    = (const float*)d_in[7];
    const float* iop    = (const float*)d_in[8];
    const float* Wq = (const float*)d_in[9],  *bq = (const float*)d_in[10];
    const float* Wk = (const float*)d_in[11], *bk = (const float*)d_in[12];
    const float* Wv = (const float*)d_in[13], *bv = (const float*)d_in[14];
    const float* Wo = (const float*)d_in[15], *bo = (const float*)d_in[16];
    const float* lnag = (const float*)d_in[17], *lnab = (const float*)d_in[18];
    const float* Wi = (const float*)d_in[19], *bi = (const float*)d_in[20];
    const float* Wf = (const float*)d_in[21], *bf = (const float*)d_in[22];
    const float* lnfg = (const float*)d_in[23], *lnfb = (const float*)d_in[24];
    const float* Wagg = (const float*)d_in[25], *bagg = (const float*)d_in[26];
    const float* Wd1 = (const float*)d_in[27], *bd1 = (const float*)d_in[28];
    const float* ln1g = (const float*)d_in[29], *ln1b = (const float*)d_in[30];
    const float* Wd2 = (const float*)d_in[31], *bd2 = (const float*)d_in[32];
    const float* ln2g = (const float*)d_in[33], *ln2b = (const float*)d_in[34];
    float* out = (float*)d_out;

    // workspace layout (floats)
    float* ws = (float*)d_ws;
    size_t off = 0;
    auto alloc = [&](size_t n) { float* p = ws + off; off += n; return p; };
    float* Uop = alloc((size_t)B_ * N_ * N_);
    float* Iop = alloc((size_t)B_ * N_ * N_);
    float* tq0 = alloc((size_t)AN * E_);
    float* tq1 = alloc((size_t)AN * E_);
    float* tq2 = alloc((size_t)AN * E_);
    float* tk0 = alloc((size_t)AN * E_);
    float* tk1 = alloc((size_t)AN * E_);
    float* tk2 = alloc((size_t)AN * E_);
    float* p1  = alloc((size_t)AN * E_);
    float* p2  = alloc((size_t)AN * E_);
    float* q    = alloc((size_t)MR * E_);
    float* kbuf = alloc((size_t)MR * E_);
    float* vbuf = alloc((size_t)MR * E_);
    float* ctx  = alloc((size_t)MR * E_);
    float* fbuf = q;                      // alias: q..ctx span == MR*FF exactly
    float* atmp = alloc((size_t)MR * E_);
    float* hs1  = alloc((size_t)MR * E_);
    float* hs2  = alloc((size_t)MR * E_);

    auto gemm = [&](const float* Am, const float* Bm, const float* bias,
                    const float* Res, float rs, float* Cm,
                    int M, int N, int K, int batch,
                    long long sAm, long long sBm, long long sCm, int act) {
        dim3 grid((unsigned)((M + 127) / 128), (unsigned)(N / 64), (unsigned)batch);
        if (act == 1)
            gemm_bf16_kernel<1><<<grid, 256, 0, stream>>>(Am, Bm, bias, Res, rs, Cm,
                                                          M, N, K, sAm, sBm, sCm);
        else
            gemm_bf16_kernel<0><<<grid, 256, 0, stream>>>(Am, Bm, bias, Res, rs, Cm,
                                                          M, N, K, sAm, sBm, sCm);
    };

    sparse_prop_kernel<<<1024, 256, 0, stream>>>(Uop, uop);
    sparse_prop_kernel<<<1024, 256, 0, stream>>>(Iop, iop);

    for (int i = 0; i < L_; ++i) {
        const float* hsIn   = (i == 0) ? hidden : hs1;
        float*       hsOut  = (i == 0) ? hs1 : hs2;
        const float* prevHs = (i == 0) ? nullptr : hs1;

        // prop inputs (must read hs1 rows 0/1 BEFORE embedding overwrite)
        prop_init_kernel<<<AN, 256, 0, stream>>>(tq0, qe, prevHs, 0);
        prop_init_kernel<<<AN, 256, 0, stream>>>(tk0, ke, prevHs, 1);

        // _prop: t = P@t + DELTA*enc, twice (batched 64x256x64 WMMA GEMMs)
        const long long sP = 64 * 64, sT = 64 * 256;
        gemm(Uop, tq0, nullptr, tq0, 0.1f, tq1, 64, 256, 64, B_, sP, sT, sT, 0);
        gemm(Uop, tq1, nullptr, tq0, 0.1f, tq2, 64, 256, 64, B_, sP, sT, sT, 0);
        gemm(Iop, tk0, nullptr, tk0, 0.1f, tk1, 64, 256, 64, B_, sP, sT, sT, 0);
        gemm(Iop, tk1, nullptr, tk0, 0.1f, tk2, 64, 256, 64, B_, sP, sT, sT, 0);

        // p1/p2 = LN(prop @ W_d + b_d)
        gemm(tq2, Wd1, bd1, nullptr, 0.f, p1, AN, 256, 256, 1, 0, 0, 0, 0);
        ln_kernel<<<AN / 8, 256, 0, stream>>>(p1, ln1g, ln1b, AN);
        gemm(tk2, Wd2, bd2, nullptr, 0.f, p2, AN, 256, 256, 1, 0, 0, 0, 0);
        ln_kernel<<<AN / 8, 256, 0, stream>>>(p2, ln2g, ln2b, AN);

        // embedding overwrite (tokens 0..3), in place on hs1 for layer 1
        if (i > 0) embed_set_kernel<<<AN, 256, 0, stream>>>(hs1, qe, ke, de, ce);

        const float* Wq_i = Wq + (size_t)i * E_ * E_, *bq_i = bq + (size_t)i * E_;
        const float* Wk_i = Wk + (size_t)i * E_ * E_, *bk_i = bk + (size_t)i * E_;
        const float* Wv_i = Wv + (size_t)i * E_ * E_, *bv_i = bv + (size_t)i * E_;
        const float* Wo_i = Wo + (size_t)i * E_ * E_, *bo_i = bo + (size_t)i * E_;
        const float* Wi_i = Wi + (size_t)i * E_ * FF_, *bi_i = bi + (size_t)i * FF_;
        const float* Wf_i = Wf + (size_t)i * FF_ * E_, *bf_i = bf + (size_t)i * E_;

        // Q/K/V projections
        gemm(hsIn, Wq_i, bq_i, nullptr, 0.f, q,    MR, 256, 256, 1, 0, 0, 0, 0);
        gemm(hsIn, Wk_i, bk_i, nullptr, 0.f, kbuf, MR, 256, 256, 1, 0, 0, 0, 0);
        gemm(hsIn, Wv_i, bv_i, nullptr, 0.f, vbuf, MR, 256, 256, 1, 0, 0, 0, 0);

        // fused attention (softmax in LDS)
        attn_kernel<<<AN * 4, 256, 0, stream>>>(q, kbuf, vbuf, amask, ctx,
                                                (i == 0) ? 1 : 0);

        // out proj + residual, LN
        gemm(ctx, Wo_i, bo_i, hsIn, 1.f, atmp, MR, 256, 256, 1, 0, 0, 0, 0);
        ln_kernel<<<MR / 8, 256, 0, stream>>>(atmp, lnag + (size_t)i * E_,
                                              lnab + (size_t)i * E_, MR);

        // FFN: gelu(a@Wi+bi) @ Wf + bf + a, LN
        gemm(atmp, Wi_i, bi_i, nullptr, 0.f, fbuf, MR, 1024, 256, 1, 0, 0, 0, 1);
        gemm(fbuf, Wf_i, bf_i, atmp, 1.f, hsOut, MR, 256, 1024, 1, 0, 0, 0, 0);
        ln_kernel<<<MR / 8, 256, 0, stream>>>(hsOut, lnfg + (size_t)i * E_,
                                              lnfb + (size_t)i * E_, MR);

        // aggregation write into hsOut[:,0,4,:]
        agg_kernel<<<B_, 256, 0, stream>>>(hsOut, p1, p2,
                                           Wagg + (size_t)i * 768 * 256,
                                           bagg + (size_t)i * 256);
    }

    final_out_kernel<<<B_, 256, 0, stream>>>(hs1, hs2, out);
}
